// HybridQLSTM_65481071407963
// MI455X (gfx1250) — matrix operand order repaired and checked by reading, not compiled, not run on hardware
//
#include <hip/hip_runtime.h>
#include <cmath>

typedef __attribute__((ext_vector_type(16))) _Float16 v16h;
typedef __attribute__((ext_vector_type(8)))  _Float16 v8h;
typedef __attribute__((ext_vector_type(8)))  float    v8f;

#define B_   256
#define S_   512
#define D_   128
#define H_   128
#define QH_  16
#define GI_  256
#define N4H  512   // 4*H stacked gate columns

// ---------------------------------------------------------------------------
// Prep 1: WxT[n][k] = lin_W[g][k][h]  (f16, transposed for WMMA B-fragments)
//         WhSum[n]  = sum_k lin_W[g][128+k][h]   (h-recurrence collapses to
//         h_scalar * colsum because h is uniform across H given zero init)
// ---------------------------------------------------------------------------
__global__ __launch_bounds__(128)
void prep_w(const float* __restrict__ linW,
            _Float16* __restrict__ WxT,
            float* __restrict__ WhSum) {
    __shared__ float red[128];
    const int n = blockIdx.x;      // 0..511 : column  n = g*128 + h
    const int k = threadIdx.x;     // 0..127 : K index
    const int g = n >> 7, h = n & 127;
    const float wx = linW[((g * GI_) + k) * H_ + h];
    const float wh = linW[((g * GI_) + 128 + k) * H_ + h];
    WxT[n * 128 + k] = (_Float16)wx;
    red[k] = wh;
    __syncthreads();
    for (int s = 64; s > 0; s >>= 1) {
        if (k < s) red[k] += red[k + s];
        __syncthreads();
    }
    if (k == 0) WhSum[n] = red[0];
}

// Prep 2: qW1T[g][q][k] = q_W1[g][k][q]  (f16)
__global__ __launch_bounds__(128)
void prep_q(const float* __restrict__ qW1, _Float16* __restrict__ qW1T) {
    const int gq = blockIdx.x;     // g*16 + q
    const int h  = threadIdx.x;    // 0..127
    const int g = gq >> 4, q = gq & 15;
    qW1T[gq * 128 + h] = (_Float16)qW1[(g * H_ + h) * QH_ + q];
}

// ---------------------------------------------------------------------------
// Big parallel GEMM: xz[b*S+t][n] = seq[b][t][:] @ WxT[n][:] + lin_b[n]
// M = 131072, K = 128, N = 512.  One wave per 16x64 tile (16 WMMAs/wave);
// all 8 waves of a block share one M-tile so A hits WGP$/L0.
// ---------------------------------------------------------------------------
__global__ __launch_bounds__(256)
void xz_gemm(const float* __restrict__ seq,
             const _Float16* __restrict__ WxT,
             const float* __restrict__ linb,
             _Float16* __restrict__ xz) {
    const int wave = threadIdx.x >> 5;
    const int lane = threadIdx.x & 31;
    const int job   = blockIdx.x * 8 + wave;
    const int mtile = job >> 3;        // 0..8191
    const int ngrp  = job & 7;         // 0..7  (4 n-tiles each)
    const int ml = lane & 15;
    const int hi = lane >> 4;
    const long row = (long)mtile * 16 + ml;

    v8f c[4] = {};
    const float* arow = seq + row * (long)D_;
    #pragma unroll
    for (int kk = 0; kk < 4; ++kk) {
        const int K0 = kk * 32;
        // A fragment layout (16-bit A 16x32): lane holds M = lane%16,
        // elements j<8 -> K = K0 + 8*hi + j ; j>=8 -> K = K0 + 16 + 8*hi + (j-8)
        const float* p0 = arow + K0 + 8 * hi;
        const float* p1 = arow + K0 + 16 + 8 * hi;
        v16h a;
        #pragma unroll
        for (int j = 0; j < 8; ++j) a[j]     = (_Float16)p0[j];
        #pragma unroll
        for (int j = 0; j < 8; ++j) a[8 + j] = (_Float16)p1[j];
        #pragma unroll
        for (int nt = 0; nt < 4; ++nt) {
            const int n = (ngrp * 4 + nt) * 16 + ml;
            // B fragment: lane holds N = lane%16, j<8 -> K = K0 + 8*hi + j ; j>=8 -> K = K0 + 16 + 8*hi + (j-8)
            const v8h b0 = *(const v8h*)(WxT + n * 128 + K0 + 8 * hi);
            const v8h b1 = *(const v8h*)(WxT + n * 128 + K0 + 16 + 8 * hi);
            const v16h b = __builtin_shufflevector(b0, b1, 0, 1, 2, 3, 4, 5, 6, 7, 8, 9, 10, 11, 12, 13, 14, 15);
            c[nt] = __builtin_amdgcn_wmma_f32_16x16x32_f16(
                false, a, false, b, (short)0, c[nt], false, false);
        }
    }
    // C layout: vgpr r -> M = r + 8*hi, N = lane%16
    #pragma unroll
    for (int nt = 0; nt < 4; ++nt) {
        const int n = (ngrp * 4 + nt) * 16 + ml;
        const float bias = linb[n];
        #pragma unroll
        for (int r = 0; r < 8; ++r) {
            const long m = (long)mtile * 16 + r + 8 * hi;
            xz[m * N4H + n] = (_Float16)(c[nt][r] + bias);
        }
    }
}

// ---------------------------------------------------------------------------
// Persistent recurrence: 16 blocks x 128 threads; block owns 16 batch rows,
// wave g owns gate g.  Per step: A = z_g[16x128] = xz + h*WhSum (f16 frags),
// 4 accumulating WMMAs against register-resident qW1T B-fragments, then
// relu -> qW2-weighted lane reduction -> tanh -> gate combine -> state update.
// WhSum runs live in registers; xz loads for step t+1 are issued while step
// t's reductions/barriers run (software pipeline) to hide HBM latency.
// ---------------------------------------------------------------------------
__global__ __launch_bounds__(128)
void recur(const _Float16* __restrict__ xz,
           const _Float16* __restrict__ qW1T,
           const float* __restrict__ WhSum,
           const float* __restrict__ qb1,
           const float* __restrict__ qW2,
           const float* __restrict__ qb2,
           float* __restrict__ out) {
    __shared__ float hs[16], cs[16];
    __shared__ float ang[4][16];

    const int tid  = threadIdx.x;
    const int g    = tid >> 5;     // wave id == gate id
    const int lane = tid & 31;
    const int ml   = lane & 15;
    const int hi   = lane >> 4;
    const int b0   = blockIdx.x * 16;

    if (tid < 16) { hs[tid] = 0.f; cs[tid] = 0.f; }

    // ---- loop-invariant per-wave data, all register resident ----
    const float qw2l = qW2[g * 16 + ml];
    const float qb1l = qb1[g * 16 + ml];
    const float qb2g = qb2[g];
    v16h bq[4];                       // qW1T B-fragments (32 VGPRs)
    float w0r[4][8], w1r[4][8];       // WhSum K-runs      (64 VGPRs)
    #pragma unroll
    for (int kk = 0; kk < 4; ++kk) {
        {
            const v8h q0 = *(const v8h*)(qW1T + (g * 16 + ml) * 128 + kk * 32 + 8 * hi);
            const v8h q1 = *(const v8h*)(qW1T + (g * 16 + ml) * 128 + kk * 32 + 16 + 8 * hi);
            bq[kk] = __builtin_shufflevector(q0, q1, 0, 1, 2, 3, 4, 5, 6, 7, 8, 9, 10, 11, 12, 13, 14, 15);
        }
        #pragma unroll
        for (int j = 0; j < 8; ++j) {
            w0r[kk][j] = WhSum[g * 128 + kk * 32 + 8 * hi + j];
            w1r[kk][j] = WhSum[g * 128 + kk * 32 + 16 + 8 * hi + j];
        }
    }

    const long rowbase = (long)(b0 + ml) * S_;
    const _Float16* xbase = xz + rowbase * (long)N4H + g * 128;

    // prologue: load step-0 xz runs
    v8h p0[4], p1[4];
    #pragma unroll
    for (int kk = 0; kk < 4; ++kk) {
        p0[kk] = *(const v8h*)(xbase + kk * 32 + 8 * hi);
        p1[kk] = *(const v8h*)(xbase + kk * 32 + 16 + 8 * hi);
    }
    __syncthreads();

    for (int t = 0; t < S_; ++t) {
        const float hm = hs[ml];
        // consume this step's prefetched xz
        v8h c0[4], c1[4];
        #pragma unroll
        for (int kk = 0; kk < 4; ++kk) { c0[kk] = p0[kk]; c1[kk] = p1[kk]; }
        // issue next step's loads early (independent of h) to hide latency
        if (t + 1 < S_) {
            const _Float16* xn = xbase + (long)(t + 1) * N4H;
            #pragma unroll
            for (int kk = 0; kk < 4; ++kk) {
                p0[kk] = *(const v8h*)(xn + kk * 32 + 8 * hi);
                p1[kk] = *(const v8h*)(xn + kk * 32 + 16 + 8 * hi);
            }
        }
        v8f acc = {};
        #pragma unroll
        for (int kk = 0; kk < 4; ++kk) {
            v16h a;
            #pragma unroll
            for (int j = 0; j < 8; ++j)
                a[j]     = (_Float16)((float)c0[kk][j] + hm * w0r[kk][j]);
            #pragma unroll
            for (int j = 0; j < 8; ++j)
                a[8 + j] = (_Float16)((float)c1[kk][j] + hm * w1r[kk][j]);
            acc = __builtin_amdgcn_wmma_f32_16x16x32_f16(
                false, a, false, bq[kk], (short)0, acc, false, false);
        }
        // relu(a + b1) * w2, reduce over q (=16 lanes within each half)
        float s[8];
        #pragma unroll
        for (int r = 0; r < 8; ++r) {
            float v = acc[r] + qb1l;
            v = v > 0.f ? v : 0.f;
            s[r] = v * qw2l;
        }
        #pragma unroll
        for (int msk = 1; msk < 16; msk <<= 1) {
            #pragma unroll
            for (int r = 0; r < 8; ++r) s[r] += __shfl_xor(s[r], msk, 32);
        }
        if (ml == 0) {
            #pragma unroll
            for (int r = 0; r < 8; ++r)
                ang[g][r + 8 * hi] = tanhf(s[r] + qb2g);
        }
        __syncthreads();
        if (tid < 16) {
            const float f = 1.f / (1.f + __expf(-ang[0][tid]));
            const float i = 1.f / (1.f + __expf(-ang[1][tid]));
            const float u = tanhf(ang[2][tid]);
            const float o = 1.f / (1.f + __expf(-ang[3][tid]));
            const float cn = f * cs[tid] + i * u;
            cs[tid] = cn;
            hs[tid] = o * tanhf(cn);
        }
        __syncthreads();
        // out[b0+m][t][0..127] = hs[m]  (uniform across H), coalesced float4
        const int m    = tid >> 3;
        const int col0 = (tid & 7) * 16;
        const float hv = hs[m];
        const float4 hv4 = make_float4(hv, hv, hv, hv);
        float4* op = (float4*)(out + ((long)(b0 + m) * S_ + t) * (long)H_ + col0);
        op[0] = hv4; op[1] = hv4; op[2] = hv4; op[3] = hv4;
    }
}

// ---------------------------------------------------------------------------
extern "C" void kernel_launch(void* const* d_in, const int* in_sizes, int n_in,
                              void* d_out, int out_size, void* d_ws, size_t ws_size,
                              hipStream_t stream) {
    const float* seq  = (const float*)d_in[0];   // [B,S,D]
    const float* linW = (const float*)d_in[1];   // [4,GI,H]
    const float* linb = (const float*)d_in[2];   // [4,H]
    const float* qW1  = (const float*)d_in[3];   // [4,H,QH]
    const float* qb1  = (const float*)d_in[4];   // [4,QH]
    const float* qW2  = (const float*)d_in[5];   // [4,QH,1]
    const float* qb2  = (const float*)d_in[6];   // [4,1]
    float* out = (float*)d_out;                  // [B,S,H]

    char* ws = (char*)d_ws;
    _Float16* xz    = (_Float16*)ws;                                   // 128 MB
    _Float16* WxT   = (_Float16*)(ws + 134217728);                     // 128 KB
    _Float16* qW1T  = (_Float16*)(ws + 134217728 + 131072);            // 16 KB
    float*    WhSum = (float*)   (ws + 134217728 + 131072 + 16384);    // 2 KB

    prep_w <<<512,  128, 0, stream>>>(linW, WxT, WhSum);
    prep_q <<<64,   128, 0, stream>>>(qW1, qW1T);
    xz_gemm<<<8192, 256, 0, stream>>>(seq, WxT, linb, xz);
    recur  <<<16,   128, 0, stream>>>(xz, qW1T, WhSum, qb1, qW2, qb2, out);
}
